// SpanAttention_64510408786370
// MI455X (gfx1250) — matrix operand order repaired and checked by reading, not compiled, not run on hardware
//
#include <hip/hip_runtime.h>

// Problem constants (match reference)
#define B_  2
#define L_  512
#define D_  768
#define MAXW 12
#define WE_ 128
#define S_  (L_ * MAXW)      // 6144 spans per batch
#define KDIM (D_ + WE_)      // 896 (GEMM K)
#define M_  (B_ * S_)        // 12288 (GEMM M)

typedef __attribute__((ext_vector_type(4)))  float  f4;
typedef __attribute__((ext_vector_type(8)))  float  v8f;
typedef __attribute__((ext_vector_type(16))) __bf16 v16bf;

typedef unsigned short ushort_t;

__device__ __forceinline__ ushort_t f2bf(float x) {           // RNE f32 -> bf16 bits
    unsigned u = __float_as_uint(x);
    u += 0x7fffu + ((u >> 16) & 1u);
    return (ushort_t)(u >> 16);
}
__device__ __forceinline__ float bf2f(ushort_t b) {
    return __uint_as_float(((unsigned)b) << 16);
}

// ---------------------------------------------------------------------------
// Kernel 1: logits[b,l] = dot(h[b,l,:], w_att) + b_att.  One wave32 per row.
// ---------------------------------------------------------------------------
__global__ void span_logits_kernel(const float* __restrict__ h,
                                   const float* __restrict__ w_att,
                                   const float* __restrict__ b_att,
                                   float* __restrict__ logits) {
    int row  = blockIdx.x * 8 + (threadIdx.x >> 5);   // 8 waves / block
    int lane = threadIdx.x & 31;
    const float* hp = h + (size_t)row * D_;
    float s = 0.f;
    #pragma unroll 4
    for (int d = lane; d < D_; d += 32) s += hp[d] * w_att[d];
    #pragma unroll
    for (int off = 16; off; off >>= 1) s += __shfl_xor(s, off, 32);
    if (lane == 0) logits[row] = s + b_att[0];
}

// ---------------------------------------------------------------------------
// Kernel 2: per-span masked softmax pooling + width-emb concat.
// Writes rep split into bf16 hi/lo planes: rep = hi + lo (split-bf16 GEMM).
// rep[row, 0:896] = [ span_rep(768) | width_emb(128) ].  One block per span.
// ---------------------------------------------------------------------------
__global__ void span_pool_kernel(const float* __restrict__ h,
                                 const int*   __restrict__ span_idx,
                                 const float* __restrict__ logits,
                                 const float* __restrict__ width_emb,
                                 ushort_t* __restrict__ rep_hi,
                                 ushort_t* __restrict__ rep_lo) {
    int row = blockIdx.x;                // b*S + s
    int b   = row / S_;
    int start = span_idx[row * 2 + 0];
    int end   = span_idx[row * 2 + 1];   // inclusive
    int wd    = end - start;             // width index, 0..11

    float att[MAXW];
    int   idxc[MAXW];
    float mx = -1e30f;
    #pragma unroll
    for (int w = 0; w < MAXW; ++w) {
        int idx  = start + w;
        bool ok  = (w <= wd) && (idx < L_);
        int ic   = idx < (L_ - 1) ? idx : (L_ - 1);
        idxc[w]  = ic;
        float lg = ok ? logits[b * L_ + ic] : -1e30f;
        att[w]   = lg;
        mx = fmaxf(mx, lg);
    }
    float ssum = 0.f;
    #pragma unroll
    for (int w = 0; w < MAXW; ++w) { att[w] = __expf(att[w] - mx); ssum += att[w]; }
    float inv = 1.f / ssum;
    #pragma unroll
    for (int w = 0; w < MAXW; ++w) att[w] *= inv;

    const float* hb = h + (size_t)b * L_ * D_;
    size_t rbase = (size_t)row * KDIM;
    for (int d = threadIdx.x; d < KDIM; d += blockDim.x) {
        float v;
        if (d < D_) {
            float acc = 0.f;
            #pragma unroll
            for (int w = 0; w < MAXW; ++w) acc += att[w] * hb[(size_t)idxc[w] * D_ + d];
            v = acc;
        } else {
            v = width_emb[wd * WE_ + (d - D_)];
        }
        ushort_t hbits = f2bf(v);
        float    resid = v - bf2f(hbits);
        rep_hi[rbase + d] = hbits;
        rep_lo[rbase + d] = f2bf(resid);
    }
}

// ---------------------------------------------------------------------------
// Kernel 3: split W_down [K,N] f32 -> transposed bf16 planes Wt_hi/Wt_lo [N,K].
// [N,K] layout makes each lane's B fragment two contiguous 16B runs.
// ---------------------------------------------------------------------------
__global__ void wdown_split_kernel(const float* __restrict__ W,
                                   ushort_t* __restrict__ wt_hi,
                                   ushort_t* __restrict__ wt_lo) {
    int k = blockIdx.x;                         // 0..KDIM-1
    for (int n = threadIdx.x; n < D_; n += blockDim.x) {
        float v = W[(size_t)k * D_ + n];
        ushort_t hbits = f2bf(v);
        float    resid = v - bf2f(hbits);
        wt_hi[(size_t)n * KDIM + k] = hbits;
        wt_lo[(size_t)n * KDIM + k] = f2bf(resid);
    }
}

// ---------------------------------------------------------------------------
// Kernel 4: out = relu(rep @ W_down + b_down) via V_WMMA_F32_16X16X32_BF16,
// 3-term split-bf16: C += Ahi*Bhi + Ahi*Blo + Alo*Bhi  (f32 accumulation).
// Block: 256 thr = 8 waves, tile 64(M) x 128(N); per-wave 32x32 (2x2 subtiles).
// Double-buffered LDS (one barrier/iter): global prefetch -> regs overlaps the
// 12 WMMAs; ds_store lands in the alternate buffer after compute.
// ---------------------------------------------------------------------------
#define GBM 64
#define GBN 128
#define GBK 32
#define LDT 40    // bf16 row stride in LDS (32 + 16B pad)

__global__ __launch_bounds__(256)
void gemm_bf16x3_wmma_kernel(const ushort_t* __restrict__ Ahi,   // rep_hi [M_,KDIM]
                             const ushort_t* __restrict__ Alo,   // rep_lo [M_,KDIM]
                             const ushort_t* __restrict__ Bhi,   // wt_hi  [D_,KDIM]
                             const ushort_t* __restrict__ Blo,   // wt_lo  [D_,KDIM]
                             const float*    __restrict__ bias,  // b_down [D_]
                             float*          __restrict__ out) { // [M_,D_]
    __shared__ ushort_t sAh[2][GBM * LDT];
    __shared__ ushort_t sAl[2][GBM * LDT];
    __shared__ ushort_t sBh[2][GBN * LDT];
    __shared__ ushort_t sBl[2][GBN * LDT];

    const int tid  = threadIdx.x;
    const int lane = tid & 31;
    const int wid  = tid >> 5;             // 0..7
    const int mt   = (wid & 1) * 32;       // wave M offset in block tile
    const int nt   = (wid >> 1) * 32;      // wave N offset in block tile
    const int mb   = blockIdx.y * GBM;
    const int nb   = blockIdx.x * GBN;
    const int lm   = lane & 15;
    const int half = lane >> 4;            // 0 or 1

    // staging coordinates: 16B chunk per thread (A: 256 chunks, B: 512 chunks)
    const int ra   = tid >> 2;             // 0..63
    const int ca   = (tid & 3) * 8;        // bf16 units
    const int rb1  = (tid + 256) >> 2;     // 64..127
    const int cb1  = ((tid + 256) & 3) * 8;

    f4 pAh, pAl, pBh0, pBl0, pBh1, pBl1;   // prefetch registers

    auto gload = [&](int k0) {
        pAh  = *(const f4*)(Ahi + (size_t)(mb + ra) * KDIM + k0 + ca);
        pAl  = *(const f4*)(Alo + (size_t)(mb + ra) * KDIM + k0 + ca);
        pBh0 = *(const f4*)(Bhi + (size_t)(nb + ra) * KDIM + k0 + ca);
        pBl0 = *(const f4*)(Blo + (size_t)(nb + ra) * KDIM + k0 + ca);
        pBh1 = *(const f4*)(Bhi + (size_t)(nb + rb1) * KDIM + k0 + cb1);
        pBl1 = *(const f4*)(Blo + (size_t)(nb + rb1) * KDIM + k0 + cb1);
    };
    auto sstore = [&](int buf) {
        *(f4*)&sAh[buf][ra * LDT + ca]   = pAh;
        *(f4*)&sAl[buf][ra * LDT + ca]   = pAl;
        *(f4*)&sBh[buf][ra * LDT + ca]   = pBh0;
        *(f4*)&sBl[buf][ra * LDT + ca]   = pBl0;
        *(f4*)&sBh[buf][rb1 * LDT + cb1] = pBh1;
        *(f4*)&sBl[buf][rb1 * LDT + cb1] = pBl1;
    };

    v8f c[2][2];
    #pragma unroll
    for (int mi = 0; mi < 2; ++mi)
        #pragma unroll
        for (int ni = 0; ni < 2; ++ni)
            c[mi][ni] = (v8f){0.f,0.f,0.f,0.f,0.f,0.f,0.f,0.f};

    gload(0);
    sstore(0);

    for (int k0 = 0, it = 0; k0 < KDIM; k0 += GBK, ++it) {
        const int  buf      = it & 1;
        const bool has_next = (k0 + GBK) < KDIM;
        if (has_next) gload(k0 + GBK);     // global latency overlaps the WMMAs

        __syncthreads();                   // publish ds_stores of this buffer

        // Load fragments (two contiguous b128 per fragment per lane).
        // A 16x32 bf16 layout: lane half h, row m: K = 8h+0..7 then 16+8h+0..7
        // B 32x16 bf16 layout: lane half h, col n: K = 8h+0..7 then 16+8h+0..7
        v16bf ah[2], al[2], bh[2], bl[2];
        #pragma unroll
        for (int mi = 0; mi < 2; ++mi) {
            const ushort_t* ph = &sAh[buf][(mt + mi * 16 + lm) * LDT];
            const ushort_t* pl = &sAl[buf][(mt + mi * 16 + lm) * LDT];
            union { v16bf v; f4 q[2]; } u;
            u.q[0] = *(const f4*)(ph + 8 * half);
            u.q[1] = *(const f4*)(ph + 16 + 8 * half);
            ah[mi] = u.v;
            u.q[0] = *(const f4*)(pl + 8 * half);
            u.q[1] = *(const f4*)(pl + 16 + 8 * half);
            al[mi] = u.v;
        }
        #pragma unroll
        for (int ni = 0; ni < 2; ++ni) {
            const ushort_t* ph = &sBh[buf][(nt + ni * 16 + lm) * LDT];
            const ushort_t* pl = &sBl[buf][(nt + ni * 16 + lm) * LDT];
            union { v16bf v; f4 q[2]; } u;
            u.q[0] = *(const f4*)(ph + 8 * half);
            u.q[1] = *(const f4*)(ph + 16 + 8 * half);
            bh[ni] = u.v;
            u.q[0] = *(const f4*)(pl + 8 * half);
            u.q[1] = *(const f4*)(pl + 16 + 8 * half);
            bl[ni] = u.v;
        }

        // 2x2 subtiles x 3 split terms = 12 WMMA per K-chunk per wave
        #pragma unroll
        for (int mi = 0; mi < 2; ++mi) {
            #pragma unroll
            for (int ni = 0; ni < 2; ++ni) {
                c[mi][ni] = __builtin_amdgcn_wmma_f32_16x16x32_bf16(
                    false, ah[mi], false, bh[ni], (short)0, c[mi][ni], false, false);
                c[mi][ni] = __builtin_amdgcn_wmma_f32_16x16x32_bf16(
                    false, ah[mi], false, bl[ni], (short)0, c[mi][ni], false, false);
                c[mi][ni] = __builtin_amdgcn_wmma_f32_16x16x32_bf16(
                    false, al[mi], false, bh[ni], (short)0, c[mi][ni], false, false);
            }
        }

        // Fill the alternate buffer; it was last read in the previous iteration, and the
        // barrier at the top of this iteration already ordered those reads.
        if (has_next) sstore(buf ^ 1);
    }

    // Epilogue: C/D layout — VGPR r: lanes 0-15 -> M=r, lanes 16-31 -> M=r+8
    #pragma unroll
    for (int mi = 0; mi < 2; ++mi) {
        int rowb = mb + mt + mi * 16 + half * 8;
        #pragma unroll
        for (int ni = 0; ni < 2; ++ni) {
            int col = nb + nt + ni * 16 + lm;
            float bv = bias[col];
            #pragma unroll
            for (int r = 0; r < 8; ++r) {
                float v = c[mi][ni][r] + bv;
                out[(size_t)(rowb + r) * D_ + col] = v > 0.f ? v : 0.f;
            }
        }
    }
}

// ---------------------------------------------------------------------------
extern "C" void kernel_launch(void* const* d_in, const int* in_sizes, int n_in,
                              void* d_out, int out_size, void* d_ws, size_t ws_size,
                              hipStream_t stream) {
    const float* h         = (const float*)d_in[0];   // [B,L,D]
    const int*   span_idx  = (const int*)  d_in[1];   // [B,S,2]
    const float* w_att     = (const float*)d_in[2];   // [D]
    const float* b_att     = (const float*)d_in[3];   // scalar
    const float* width_emb = (const float*)d_in[4];   // [MAXW,WE]
    const float* W_down    = (const float*)d_in[5];   // [KDIM,D]
    const float* b_down    = (const float*)d_in[6];   // [D]
    float*       out       = (float*)d_out;           // [B,L,MAXW,D] == [M_,D_]

    // Workspace layout (all 16B-aligned offsets):
    //   logits  : B_*L_ f32                 (4 KB)
    //   rep_hi  : M_*KDIM bf16              (22 MB)
    //   rep_lo  : M_*KDIM bf16              (22 MB)
    //   wt_hi   : D_*KDIM bf16              (1.375 MB)
    //   wt_lo   : D_*KDIM bf16              (1.375 MB)
    float*    logits = (float*)d_ws;
    ushort_t* rep_hi = (ushort_t*)(logits + B_ * L_);
    ushort_t* rep_lo = rep_hi + (size_t)M_ * KDIM;
    ushort_t* wt_hi  = rep_lo + (size_t)M_ * KDIM;
    ushort_t* wt_lo  = wt_hi + (size_t)D_ * KDIM;

    span_logits_kernel<<<(B_ * L_) / 8, 256, 0, stream>>>(h, w_att, b_att, logits);
    span_pool_kernel<<<M_, 128, 0, stream>>>(h, span_idx, logits, width_emb,
                                             rep_hi, rep_lo);
    wdown_split_kernel<<<KDIM, 256, 0, stream>>>(W_down, wt_hi, wt_lo);

    dim3 grid(D_ / GBN, M_ / GBM);   // (6, 192)
    gemm_bf16x3_wmma_kernel<<<grid, 256, 0, stream>>>(rep_hi, rep_lo, wt_hi, wt_lo,
                                                      b_down, out);
}
